// CausalSelfAttention_74603581931810
// MI455X (gfx1250) — matrix operand order repaired and checked by reading, not compiled, not run on hardware
//
#include <hip/hip_runtime.h>
#include <math.h>

// ---------------------------------------------------------------------------
// CausalSelfAttention forward, MI455X (gfx1250), wave32 + WMMA bf16.
//   B=2 T=2048 C=1024 H=16 D=64
// Pipeline: cvt(x) / transpose-cvt(W) -> QKV WMMA GEMM (+scatter to heads)
//           -> flash attention (WMMA QK^T, online softmax, WMMA P*V)
//           -> proj WMMA GEMM -> fp32 out.
// Tiles staged with GLOBAL_LOAD_ASYNC_TO_LDS_B128 (ASYNCcnt), double-buffered.
// ---------------------------------------------------------------------------

#define B_   2
#define T_   2048
#define C_   1024
#define H_   16
#define D_   64
#define NTOK (B_ * T_)   // 4096
#define N3C  (3 * C_)    // 3072
#define LDT  40          // LDS row stride (bf16 elems) for 32-wide K tiles

typedef __attribute__((ext_vector_type(16))) __bf16 v16bf;
typedef __attribute__((ext_vector_type(8)))  float  v8f;

struct alignas(16) Chunk8 { __bf16 h[8]; };          // 16-byte bf16 chunk
union FragBF { v16bf v; Chunk8 c[2]; };              // one WMMA 16-bf16 operand

__device__ __forceinline__ v8f wmma_bf16(const FragBF& a, const FragBF& b, v8f c) {
  // D = A(16x32) * B(32x16) + C, f32 accum
  return __builtin_amdgcn_wmma_f32_16x16x32_bf16(
      false, a.v, false, b.v, (short)0, c, false, false);
}

// LDS byte offset of a shared-memory pointer: per ISA aperture rule the low
// 32 bits of a flat LDS address are the LDS address.
__device__ __forceinline__ unsigned ldsoff(const void* p) {
  return (unsigned)(size_t)p;
}

// One 16B-per-lane async copy global -> LDS (tracked by ASYNCcnt).
__device__ __forceinline__ void async_copy16(unsigned lds_addr, const void* gaddr) {
  asm volatile("global_load_async_to_lds_b128 %0, %1, off"
               :: "v"(lds_addr), "v"(gaddr)
               : "memory");
}
__device__ __forceinline__ void wait_async0() {
  asm volatile("s_wait_asynccnt 0" ::: "memory");
}

// all-reduce within each 16-lane half of the wave32 (rows of a C-frag)
__device__ __forceinline__ float half16_max(float v) {
#pragma unroll
  for (int m = 1; m < 16; m <<= 1) v = fmaxf(v, __shfl_xor(v, m, 32));
  return v;
}
__device__ __forceinline__ float half16_sum(float v) {
#pragma unroll
  for (int m = 1; m < 16; m <<= 1) v += __shfl_xor(v, m, 32);
  return v;
}

// ---------------------------------------------------------------------------
// Elementwise fp32 -> bf16 (x activation)
// ---------------------------------------------------------------------------
__global__ __launch_bounds__(256) void cvt_f32_bf16(
    const float* __restrict__ in, __bf16* __restrict__ out, int n) {
  int i = (blockIdx.x * 256 + threadIdx.x) * 4;
  if (i + 3 < n) {
    float4 f = *(const float4*)(in + i);
    out[i + 0] = (__bf16)f.x;
    out[i + 1] = (__bf16)f.y;
    out[i + 2] = (__bf16)f.z;
    out[i + 3] = (__bf16)f.w;
  }
}

// ---------------------------------------------------------------------------
// Transpose + convert: in[rows][cols] f32 -> out[cols][rows] bf16
// (weights stored K-contiguous so WMMA B-frags are 16B-contiguous per lane)
// ---------------------------------------------------------------------------
__global__ __launch_bounds__(256) void transpose_cvt(
    const float* __restrict__ in, __bf16* __restrict__ out, int rows, int cols) {
  __shared__ float tile[32][33];
  int c0 = blockIdx.x * 32, r0 = blockIdx.y * 32;
  int tx = threadIdx.x, ty = threadIdx.y;   // block (32,8)
#pragma unroll
  for (int i = ty; i < 32; i += 8)
    tile[i][tx] = in[(size_t)(r0 + i) * cols + (c0 + tx)];
  __syncthreads();
#pragma unroll
  for (int i = ty; i < 32; i += 8)
    out[(size_t)(c0 + i) * rows + (r0 + tx)] = (__bf16)tile[tx][i];
}

// ---------------------------------------------------------------------------
// Shared 128x128 (per block) WMMA GEMM mainloop, double-buffered async tiles.
//   A  : [M][KDIM] bf16 row-major        (activations)
//   Bt : [N][KDIM] bf16 row-major        (transposed weights)
// 8 waves: waveM = wid&1 (64 rows), waveN = wid>>1 (32 cols)
// As/Bs: 2 buffers of 128 x LDT bf16 each.
// ---------------------------------------------------------------------------
template <int KDIM>
__device__ __forceinline__ void gemm_mainloop(
    const __bf16* __restrict__ A, const __bf16* __restrict__ Bt,
    int mBase, int nBase, __bf16* As, __bf16* Bs, v8f acc[4][2]) {
  const int tid = threadIdx.x;
  const int lane = tid & 31, wid = tid >> 5;
  const int hl = lane >> 4, l16 = lane & 15;
  const int waveM = wid & 1, waveN = wid >> 1;

  // stage one 128x32 K-tile of A and Bt into LDS buffer `buf` (async)
  auto stage = [&](int kk, int buf) {
    __bf16* AsB = As + buf * (128 * LDT);
    __bf16* BsB = Bs + buf * (128 * LDT);
#pragma unroll
    for (int i = 0; i < 2; ++i) {
      int chunk = tid + i * 256;
      int r = chunk >> 2, c8 = (chunk & 3) * 8;
      async_copy16(ldsoff(&AsB[r * LDT + c8]),
                   A + (size_t)(mBase + r) * KDIM + kk + c8);
      async_copy16(ldsoff(&BsB[r * LDT + c8]),
                   Bt + (size_t)(nBase + r) * KDIM + kk + c8);
    }
  };

  stage(0, 0);
  wait_async0();
  __syncthreads();

  for (int kk = 0; kk < KDIM; kk += 32) {
    const int buf = (kk >> 5) & 1;
    if (kk + 32 < KDIM) stage(kk + 32, buf ^ 1);          // overlap with WMMAs
    if (kk + 64 < KDIM) {                                  // warm L2 one further
      __builtin_prefetch(A + (size_t)(mBase + (tid >> 1)) * KDIM + kk + 64 + (tid & 1) * 16, 0, 1);
      __builtin_prefetch(Bt + (size_t)(nBase + (tid >> 1)) * KDIM + kk + 64 + (tid & 1) * 16, 0, 1);
    }
    const __bf16* AsB = As + buf * (128 * LDT);
    const __bf16* BsB = Bs + buf * (128 * LDT);

    FragBF afr[4], bfr[2];
#pragma unroll
    for (int ms = 0; ms < 4; ++ms) {
      // A-frag: lane row = l16; chunks at k = 8*hl and 16+8*hl
      const __bf16* p = &AsB[(waveM * 64 + ms * 16 + l16) * LDT + hl * 8];
      afr[ms].c[0] = *(const Chunk8*)(p);
      afr[ms].c[1] = *(const Chunk8*)(p + 16);
    }
#pragma unroll
    for (int ns = 0; ns < 2; ++ns) {
      // B-frag: lane col = l16; chunks at k = 8*hl and 16+8*hl
      const __bf16* p = &BsB[(waveN * 32 + ns * 16 + l16) * LDT + hl * 8];
      bfr[ns].c[0] = *(const Chunk8*)(p);
      bfr[ns].c[1] = *(const Chunk8*)(p + 16);
    }
#pragma unroll
    for (int ms = 0; ms < 4; ++ms)
#pragma unroll
      for (int ns = 0; ns < 2; ++ns)
        acc[ms][ns] = wmma_bf16(afr[ms], bfr[ns], acc[ms][ns]);

    wait_async0();      // my next-tile copies have landed
    __syncthreads();    // everyone done reading buf / writing buf^1
  }
}

// ---------------------------------------------------------------------------
// QKV GEMM: [4096,1024] x [1024,3072] + bias, scattered to
//   Q [B,H,T,D], K [B,H,T,D], Vt [B,H,D,T]  (bf16)
// ---------------------------------------------------------------------------
__global__ __launch_bounds__(256) void qkv_gemm_scatter(
    const __bf16* __restrict__ xb, const __bf16* __restrict__ wqkvt,
    const float* __restrict__ bqkv,
    __bf16* __restrict__ q, __bf16* __restrict__ k, __bf16* __restrict__ vt) {
  __shared__ __bf16 As[2 * 128 * LDT];
  __shared__ __bf16 Bs[2 * 128 * LDT];
  int mBase = blockIdx.y * 128;
  int nBase = blockIdx.x * 128;

  v8f acc[4][2];
  v8f z = {};
#pragma unroll
  for (int i = 0; i < 4; ++i)
#pragma unroll
    for (int j = 0; j < 2; ++j) acc[i][j] = z;

  gemm_mainloop<C_>(xb, wqkvt, mBase, nBase, As, Bs, acc);

  const int lane = threadIdx.x & 31, wid = threadIdx.x >> 5;
  const int hl = lane >> 4, l16 = lane & 15;
  const int waveM = wid & 1, waveN = wid >> 1;
#pragma unroll
  for (int ms = 0; ms < 4; ++ms)
#pragma unroll
    for (int ns = 0; ns < 2; ++ns)
#pragma unroll
      for (int v = 0; v < 8; ++v) {
        int M = mBase + waveM * 64 + ms * 16 + v + 8 * hl;   // token
        int N = nBase + waveN * 32 + ns * 16 + l16;          // qkv channel
        __bf16 h = (__bf16)(acc[ms][ns][v] + bqkv[N]);
        int bb = M >> 11;           // / T_
        int t  = M & (T_ - 1);
        if (N < C_) {
          int hh = N >> 6, d = N & 63;
          q[((size_t)(bb * H_ + hh) * T_ + t) * D_ + d] = h;
        } else if (N < 2 * C_) {
          int cc = N - C_;
          int hh = cc >> 6, d = cc & 63;
          k[((size_t)(bb * H_ + hh) * T_ + t) * D_ + d] = h;
        } else {
          int cc = N - 2 * C_;
          int hh = cc >> 6, d = cc & 63;
          vt[((size_t)(bb * H_ + hh) * D_ + d) * T_ + t] = h;  // transposed V
        }
      }
}

// ---------------------------------------------------------------------------
// Flash attention: block = 128 q rows, 8 waves x 16 rows; 32-key tiles.
// S = Q K^T via WMMA (d-contiguous), online softmax with half-wave reductions,
// P re-laid out C-frag -> A-frag through per-wave LDS, O += P * V^T via WMMA.
// K/V tiles double-buffered via async LDS copies.
// ---------------------------------------------------------------------------
__global__ __launch_bounds__(256) void flash_attn(
    const __bf16* __restrict__ q, const __bf16* __restrict__ k,
    const __bf16* __restrict__ vt, __bf16* __restrict__ yb) {
  __shared__ __bf16 Ks[2 * 32 * 72];    // 32 keys x 64 d (stride 72), x2 bufs
  __shared__ __bf16 Vs[2 * 64 * LDT];   // 64 d x 32 keys (stride 40), x2 bufs
  __shared__ __bf16 Ps[8 * 16 * LDT];   // per-wave 16x32 P staging

  const int bh = blockIdx.y;
  const int qTile = blockIdx.x * 128;
  const int tid = threadIdx.x, lane = tid & 31, wid = tid >> 5;
  const int hl = lane >> 4, l16 = lane & 15;
  const int qRow0 = qTile + wid * 16;

  const __bf16* qp = q + (size_t)bh * T_ * D_;
  const __bf16* kp = k + (size_t)bh * T_ * D_;
  const __bf16* vp = vt + (size_t)bh * D_ * T_;

  // stage K (32x64) and Vt (64x32) tiles for key base kv into buffer `buf`
  auto stage_kv = [&](int kv, int buf) {
    __bf16* KsB = Ks + buf * (32 * 72);
    __bf16* VsB = Vs + buf * (64 * LDT);
    {
      int r = tid >> 3, c8 = (tid & 7) * 8;
      async_copy16(ldsoff(&KsB[r * 72 + c8]),
                   kp + (size_t)(kv + r) * D_ + c8);
    }
    {
      int r = tid >> 2, c8 = (tid & 3) * 8;
      async_copy16(ldsoff(&VsB[r * LDT + c8]),
                   vp + (size_t)r * T_ + kv + c8);
    }
  };

  // Q frags live in registers for the whole block of keys
  FragBF qa[2];
  {
    const __bf16* base = qp + (size_t)(qRow0 + l16) * D_;
#pragma unroll
    for (int s = 0; s < 2; ++s) {
      int c0 = s * 32 + hl * 8;
      qa[s].c[0] = *(const Chunk8*)(base + c0);
      qa[s].c[1] = *(const Chunk8*)(base + c0 + 16);
    }
  }

  float mstat[8], lstat[8];
  v8f o[4];
  v8f z = {};
#pragma unroll
  for (int v = 0; v < 8; ++v) { mstat[v] = -1e30f; lstat[v] = 0.f; }
#pragma unroll
  for (int i = 0; i < 4; ++i) o[i] = z;

  const float sm_scale = 0.125f;  // 1/sqrt(64)
  const int kvEnd = qTile + 128;
  __bf16* pw = Ps + wid * 16 * LDT;

  stage_kv(0, 0);
  wait_async0();
  __syncthreads();

  for (int kv = 0; kv < kvEnd; kv += 32) {
    const int buf = (kv >> 5) & 1;
    if (kv + 32 < kvEnd) stage_kv(kv + 32, buf ^ 1);   // overlap with compute
    const __bf16* KsB = Ks + buf * (32 * 72);
    const __bf16* VsB = Vs + buf * (64 * LDT);

    if (kv <= qRow0 + 15) {  // wave-uniform causal tile skip (EXEC stays full)
      // ---- S = Q K^T over d = 0..63 (2 WMMA K-steps), 2 key subtiles ----
      v8f s0 = z, s1 = z;
#pragma unroll
      for (int s = 0; s < 2; ++s) {
        FragBF b0, b1;
        const __bf16* p0 = &KsB[l16 * 72 + s * 32 + hl * 8];
        const __bf16* p1 = &KsB[(16 + l16) * 72 + s * 32 + hl * 8];
        b0.c[0] = *(const Chunk8*)(p0); b0.c[1] = *(const Chunk8*)(p0 + 16);
        b1.c[0] = *(const Chunk8*)(p1); b1.c[1] = *(const Chunk8*)(p1 + 16);
        s0 = wmma_bf16(qa[s], b0, s0);
        s1 = wmma_bf16(qa[s], b1, s1);
      }

      // ---- mask + scale + online softmax (rows = v + 8*hl) ----
#pragma unroll
      for (int v = 0; v < 8; ++v) {
        int qg = qRow0 + v + 8 * hl;
        int kg0 = kv + l16, kg1 = kv + 16 + l16;
        float x0 = (kg0 <= qg) ? s0[v] * sm_scale : -1e30f;
        float x1 = (kg1 <= qg) ? s1[v] * sm_scale : -1e30f;
        float mn = fmaxf(mstat[v], half16_max(fmaxf(x0, x1)));
        float sc = __expf(mstat[v] - mn);
        float p0 = __expf(x0 - mn);
        float p1 = __expf(x1 - mn);
        lstat[v] = lstat[v] * sc + half16_sum(p0 + p1);
        mstat[v] = mn;
#pragma unroll
        for (int d4 = 0; d4 < 4; ++d4) o[d4][v] *= sc;
        int prow = v + 8 * hl;
        pw[prow * LDT + l16]      = (__bf16)p0;
        pw[prow * LDT + 16 + l16] = (__bf16)p1;
      }

      // ---- re-read P in A-frag layout (same-wave LDS, DS in-order) ----
      FragBF pa;
      {
        const __bf16* pb = pw + l16 * LDT + hl * 8;
        pa.c[0] = *(const Chunk8*)(pb);
        pa.c[1] = *(const Chunk8*)(pb + 16);
      }
      // ---- O += P (16x32) * V^T-slices (32x16 per d-subtile) ----
#pragma unroll
      for (int d4 = 0; d4 < 4; ++d4) {
        FragBF bv;
        const __bf16* vb = &VsB[(d4 * 16 + l16) * LDT + hl * 8];
        bv.c[0] = *(const Chunk8*)(vb);
        bv.c[1] = *(const Chunk8*)(vb + 16);
        o[d4] = wmma_bf16(pa, bv, o[d4]);
      }
    }
    wait_async0();
    __syncthreads();
  }

  // ---- normalize + store to y [B,T,C] bf16 ----
  const int bb = bh / H_, hh = bh % H_;
#pragma unroll
  for (int v = 0; v < 8; ++v) {
    float rl = 1.0f / lstat[v];
    int t = qRow0 + v + 8 * hl;
    __bf16* yrow = yb + ((size_t)(bb * T_ + t)) * C_ + hh * D_;
#pragma unroll
    for (int d4 = 0; d4 < 4; ++d4)
      yrow[d4 * 16 + l16] = (__bf16)(o[d4][v] * rl);
  }
}

// ---------------------------------------------------------------------------
// Output projection GEMM: [4096,1024] x [1024,1024] + bias -> fp32 out
// ---------------------------------------------------------------------------
__global__ __launch_bounds__(256) void proj_gemm(
    const __bf16* __restrict__ yb, const __bf16* __restrict__ wprot,
    const float* __restrict__ bproj, float* __restrict__ out) {
  __shared__ __bf16 As[2 * 128 * LDT];
  __shared__ __bf16 Bs[2 * 128 * LDT];
  int mBase = blockIdx.y * 128;
  int nBase = blockIdx.x * 128;

  v8f acc[4][2];
  v8f z = {};
#pragma unroll
  for (int i = 0; i < 4; ++i)
#pragma unroll
    for (int j = 0; j < 2; ++j) acc[i][j] = z;

  gemm_mainloop<C_>(yb, wprot, mBase, nBase, As, Bs, acc);

  const int lane = threadIdx.x & 31, wid = threadIdx.x >> 5;
  const int hl = lane >> 4, l16 = lane & 15;
  const int waveM = wid & 1, waveN = wid >> 1;
#pragma unroll
  for (int ms = 0; ms < 4; ++ms)
#pragma unroll
    for (int ns = 0; ns < 2; ++ns)
#pragma unroll
      for (int v = 0; v < 8; ++v) {
        int M = mBase + waveM * 64 + ms * 16 + v + 8 * hl;
        int N = nBase + waveN * 32 + ns * 16 + l16;
        out[(size_t)M * C_ + N] = acc[ms][ns][v] + bproj[N];
      }
}

// ---------------------------------------------------------------------------
extern "C" void kernel_launch(void* const* d_in, const int* in_sizes, int n_in,
                              void* d_out, int out_size, void* d_ws, size_t ws_size,
                              hipStream_t stream) {
  const float* x     = (const float*)d_in[0];
  const float* Wqkv  = (const float*)d_in[1];
  const float* bqkv  = (const float*)d_in[2];
  const float* Wproj = (const float*)d_in[3];
  const float* bproj = (const float*)d_in[4];
  float* out = (float*)d_out;

  char* ws = (char*)d_ws;
  size_t off = 0;
  auto alloc = [&](size_t bytes) -> void* {
    void* p = ws + off;
    off += (bytes + 255) & ~(size_t)255;
    return p;
  };
  __bf16* xb    = (__bf16*)alloc((size_t)NTOK * C_ * 2);
  __bf16* wqkvt = (__bf16*)alloc((size_t)N3C * C_ * 2);
  __bf16* wprot = (__bf16*)alloc((size_t)C_ * C_ * 2);
  __bf16* qb    = (__bf16*)alloc((size_t)B_ * H_ * T_ * D_ * 2);
  __bf16* kb    = (__bf16*)alloc((size_t)B_ * H_ * T_ * D_ * 2);
  __bf16* vtb   = (__bf16*)alloc((size_t)B_ * H_ * D_ * T_ * 2);
  __bf16* yb    = (__bf16*)alloc((size_t)NTOK * C_ * 2);
  (void)ws_size; (void)in_sizes; (void)n_in; (void)out_size;

  cvt_f32_bf16<<<(NTOK * C_) / (256 * 4), 256, 0, stream>>>(x, xb, NTOK * C_);
  transpose_cvt<<<dim3(N3C / 32, C_ / 32), dim3(32, 8), 0, stream>>>(Wqkv, wqkvt, C_, N3C);
  transpose_cvt<<<dim3(C_ / 32, C_ / 32), dim3(32, 8), 0, stream>>>(Wproj, wprot, C_, C_);

  qkv_gemm_scatter<<<dim3(N3C / 128, NTOK / 128), 256, 0, stream>>>(
      xb, wqkvt, bqkv, qb, kb, vtb);

  flash_attn<<<dim3(T_ / 128, B_ * H_), 256, 0, stream>>>(qb, kb, vtb, yb);

  proj_gemm<<<dim3(C_ / 128, NTOK / 128), 256, 0, stream>>>(yb, wprot, bproj, out);
}
